// ReluLayer_3977139716222
// MI455X (gfx1250) — compile-verified, matrix operand order checked by reading, not run on hardware
//
#include <hip/hip_runtime.h>
#include <hip/hip_bf16.h>

// z = ReLU(x @ W^T + b), x:[4096,4096] f32, W:[4096,4096] f32, b:[4096] f32
// bf16 WMMA (v_wmma_f32_16x16x32_bf16), f32 accumulation, BK=64,
// LDS double-buffered: global loads of stage k+1 overlap WMMA of stage k.

typedef __attribute__((ext_vector_type(16))) __bf16 v16bf;
typedef __attribute__((ext_vector_type(8)))  __bf16 bf16x8;
typedef __attribute__((ext_vector_type(4)))  __bf16 bf16x4;
typedef __attribute__((ext_vector_type(8)))  float  v8f;
typedef __attribute__((ext_vector_type(4)))  float  f32x4;

#define DDIM 4096
#define BM   128
#define BN   128
#define BK   64
#define LDSK 72              // padded row stride (bf16): 144B -> conflict-free b128 reads
#define BUFE (BM * LDSK)     // elements per buffer

union AFrag { v16bf v; bf16x8 h[2]; };

__global__ __launch_bounds__(256)
void gemm_bias_relu_bf16wmma(const float* __restrict__ X,
                             const float* __restrict__ W,
                             const float* __restrict__ Bv,
                             float* __restrict__ Out)
{
    __shared__ __bf16 Xs[2 * BUFE];
    __shared__ __bf16 Ws[2 * BUFE];

    const int tid  = threadIdx.x;
    const int lane = tid & 31;
    const int wave = tid >> 5;
    const int half = lane >> 4;          // K-half selector per ISA 16-bit layout
    const int lm   = lane & 15;          // row (A) / column (B) within 16
    const int wm   = (wave >> 1) * 32;   // wave M offset: 0,32,64,96
    const int wn   = (wave & 1)  * 64;   // wave N offset: 0,64
    const int bm   = blockIdx.y * BM;
    const int bn   = blockIdx.x * BN;

    // loader geometry: row = (tid>>4) + 16*i, col = (tid&15)*4  (i = 0..7 via immediates)
    const int lrow = tid >> 4;
    const int lc4  = (tid & 15) << 2;
    const float* Xbase = X + (size_t)(bm + lrow) * DDIM + lc4;
    const float* Wbase = W + (size_t)(bn + lrow) * DDIM + lc4;
    const int    lbase = lrow * LDSK + lc4;

    v8f acc[2][4];
#pragma unroll
    for (int mi = 0; mi < 2; ++mi)
#pragma unroll
        for (int ni = 0; ni < 4; ++ni)
            acc[mi][ni] = (v8f)0.0f;

    // ---- prologue: stage kt=0 into buffer 0 ----
#pragma unroll
    for (int i = 0; i < 8; ++i) {
        f32x4 xv = *(const f32x4*)(Xbase + i * 16 * DDIM);
        f32x4 wv = *(const f32x4*)(Wbase + i * 16 * DDIM);
        *(bf16x4*)&Xs[lbase + i * 16 * LDSK] = __builtin_convertvector(xv, bf16x4);
        *(bf16x4*)&Ws[lbase + i * 16 * LDSK] = __builtin_convertvector(wv, bf16x4);
    }
    __syncthreads();

    int bufOff = 0;
    for (int kt = 0; kt < DDIM; kt += BK) {
        const bool has_next = (kt + BK) < DDIM;
        const int  nxtOff   = bufOff ^ BUFE;

        // ---- issue next stage's global loads (in flight during WMMA) ----
        f32x4 xr[8], wr[8];
        if (has_next) {
#pragma unroll
            for (int i = 0; i < 8; ++i)
                xr[i] = *(const f32x4*)(Xbase + (kt + BK) + i * 16 * DDIM);
#pragma unroll
            for (int i = 0; i < 8; ++i)
                wr[i] = *(const f32x4*)(Wbase + (kt + BK) + i * 16 * DDIM);
            if (kt + 2 * BK < DDIM) {
                __builtin_prefetch(Xbase + kt + 2 * BK, 0, 1);
                __builtin_prefetch(Wbase + kt + 2 * BK, 0, 1);
            }
        }

        // ---- compute current stage from LDS buffer `bufOff` ----
        const __bf16* Xb = Xs + bufOff;
        const __bf16* Wb = Ws + bufOff;
#pragma unroll
        for (int kk = 0; kk < BK; kk += 32) {
            // fragment K-pattern per ISA 16-bit A layout:
            // lanes 0-15: elems 0..7 = K kk+0..7,  elems 8..15 = K kk+16..23
            // lanes 16-31: elems 0..7 = K kk+8..15, elems 8..15 = K kk+24..31
            AFrag a[2], b[4];
#pragma unroll
            for (int mi = 0; mi < 2; ++mi) {
                int r = wm + mi * 16 + lm;
                a[mi].h[0] = *(const bf16x8*)&Xb[r * LDSK + kk + 8 * half];
                a[mi].h[1] = *(const bf16x8*)&Xb[r * LDSK + kk + 16 + 8 * half];
            }
#pragma unroll
            for (int ni = 0; ni < 4; ++ni) {
                int r = wn + ni * 16 + lm;
                b[ni].h[0] = *(const bf16x8*)&Wb[r * LDSK + kk + 8 * half];
                b[ni].h[1] = *(const bf16x8*)&Wb[r * LDSK + kk + 16 + 8 * half];
            }
#pragma unroll
            for (int mi = 0; mi < 2; ++mi)
#pragma unroll
                for (int ni = 0; ni < 4; ++ni)
                    acc[mi][ni] = __builtin_amdgcn_wmma_f32_16x16x32_bf16(
                        false, a[mi].v, false, b[ni].v,
                        (short)0, acc[mi][ni], false, false);
        }

        // ---- convert + store next stage into the other buffer ----
        if (has_next) {
#pragma unroll
            for (int i = 0; i < 8; ++i) {
                *(bf16x4*)&Xs[nxtOff + lbase + i * 16 * LDSK] =
                    __builtin_convertvector(xr[i], bf16x4);
                *(bf16x4*)&Ws[nxtOff + lbase + i * 16 * LDSK] =
                    __builtin_convertvector(wr[i], bf16x4);
            }
        }
        __syncthreads();     // orders buf^1 writes vs next stage's reads
        bufOff = nxtOff;
    }

    // ---- epilogue: bias + ReLU; C layout: VGPR r -> M=r (lanes 0-15) / r+8 (16-31)
    // Output written once, never re-read -> nontemporal stores preserve L2 for X/W.
#pragma unroll
    for (int ni = 0; ni < 4; ++ni) {
        int col   = bn + wn + ni * 16 + lm;
        float bia = Bv[col];
#pragma unroll
        for (int mi = 0; mi < 2; ++mi) {
#pragma unroll
            for (int r = 0; r < 8; ++r) {
                int row = bm + wm + mi * 16 + 8 * half + r;
                float v = acc[mi][ni][r] + bia;
                __builtin_nontemporal_store(v > 0.0f ? v : 0.0f,
                                            Out + (size_t)row * DDIM + col);
            }
        }
    }
}

extern "C" void kernel_launch(void* const* d_in, const int* in_sizes, int n_in,
                              void* d_out, int out_size, void* d_ws, size_t ws_size,
                              hipStream_t stream) {
    (void)in_sizes; (void)n_in; (void)out_size; (void)d_ws; (void)ws_size;
    const float* x = (const float*)d_in[0];
    const float* W = (const float*)d_in[1];
    const float* b = (const float*)d_in[2];
    float* out = (float*)d_out;

    dim3 grid(DDIM / BN, DDIM / BM);   // 32 x 32 workgroups
    gemm_bias_relu_bf16wmma<<<grid, 256, 0, stream>>>(x, W, b, out);
}
